// ATOCActorNet_11914239279351
// MI455X (gfx1250) — compile-verified, matrix operand order checked by reading, not run on hardware
//
#include <hip/hip_runtime.h>
#include <hip/hip_bf16.h>

// Problem dims (match reference)
#define B_  1024
#define A_  64
#define N_  256
#define M_  128
#define H_  64
#define ACT_ 64
#define ROWS_ (B_ * A_)   // 65536

typedef __attribute__((ext_vector_type(16))) __bf16 v16bf;
typedef __attribute__((ext_vector_type(2)))  __bf16 v2bf;
typedef __attribute__((ext_vector_type(8)))  float  v8f;

union BF16Frag { v16bf v; unsigned int u[8]; };

#if defined(__has_builtin)
#if __has_builtin(__builtin_amdgcn_cvt_pk_bf16_f32)
#define HAVE_CVT_PK_BF16 1
#endif
#endif

__device__ __forceinline__ unsigned int bf16_1(float f) {
  unsigned int u = __float_as_uint(f);
  return ((u + 0x7FFFu + ((u >> 16) & 1u)) >> 16) & 0xFFFFu;   // RNE f32->bf16
}
__device__ __forceinline__ unsigned int bf16_pack(float lo, float hi) {
#ifdef HAVE_CVT_PK_BF16
  union { v2bf v; unsigned int u; } cv;
  cv.v = __builtin_amdgcn_cvt_pk_bf16_f32(lo, hi);   // single v_cvt_pk_bf16_f32
  return cv.u;
#else
  return bf16_1(lo) | (bf16_1(hi) << 16);
#endif
}
__device__ __forceinline__ float sigm(float x) { return 1.0f / (1.0f + __expf(-x)); }

__device__ __forceinline__ v8f vzero8() {
  v8f z;
#pragma unroll
  for (int e = 0; e < 8; ++e) z[e] = 0.0f;
  return z;
}

__device__ __forceinline__ v8f wmma_bf16(v16bf a, v16bf b, v8f c) {
  // D = A(16x32) x B(32x16) + C, f32 accumulate
  return __builtin_amdgcn_wmma_f32_16x16x32_bf16(false, a, false, b, (short)0, c, false, false);
}

// K offset for packed-pair j (0..7) per ISA 16-bit A 16x32 layout:
// VGPR j holds K = (j<4?0:16) + (j&3)*2 + 8*half, elements K and K+1.
__device__ __forceinline__ int frag_k(int kk, int j, int half) {
  return kk + ((j & 4) << 2) + ((j & 3) << 1) + (half << 3);
}

// Fragment from row-major bf16 matrix. Works for A (rows of X) and for B,
// since B[k][n] = W[n][k] for C = X @ W.T (row-major W, "base" = n-tile base).
__device__ __forceinline__ v16bf frag_bf16(const unsigned short* W, int ld, int base, int kk, int lane) {
  BF16Frag f;
  const int idx = lane & 15, half = lane >> 4;
  const unsigned short* p = W + (size_t)(base + idx) * ld;
#pragma unroll
  for (int j = 0; j < 8; ++j)
    f.u[j] = *(const unsigned int*)(p + frag_k(kk, j, half));   // 4B aligned (k even)
  return f.v;
}

// Fragment from row-major f32 matrix, converted to bf16 on the fly (optional ReLU).
__device__ __forceinline__ v16bf frag_f32(const float* X, int ld, int base, int kk, int lane, bool relu) {
  BF16Frag f;
  const int idx = lane & 15, half = lane >> 4;
  const float* p = X + (size_t)(base + idx) * ld;
#pragma unroll
  for (int j = 0; j < 8; ++j) {
    const int kb = frag_k(kk, j, half);
    float a0 = p[kb], a1 = p[kb + 1];
    if (relu) { a0 = fmaxf(a0, 0.0f); a1 = fmaxf(a1, 0.0f); }
    f.u[j] = bf16_pack(a0, a1);
  }
  return f.v;
}

// Fragment from LDS packed-bf16 tile (pitchu = uints per row, row = lane&15).
__device__ __forceinline__ v16bf frag_lds(const unsigned int* L, int pitchu, int kk, int lane) {
  BF16Frag f;
  const int idx = lane & 15, half = lane >> 4;
  const unsigned int* p = L + idx * pitchu;
#pragma unroll
  for (int j = 0; j < 8; ++j)
    f.u[j] = p[frag_k(kk, j, half) >> 1];
  return f.v;
}

// ---------------------------------------------------------------------------
// Kernel 0: f32 -> bf16 weight conversion
// ---------------------------------------------------------------------------
__global__ void k_cvt(const float* __restrict__ s, unsigned short* __restrict__ d, int n) {
  int i = blockIdx.x * blockDim.x + threadIdx.x;
  if (i < n) d[i] = (unsigned short)bf16_1(s[i]);
}

// ---------------------------------------------------------------------------
// Kernel 1: actor_1  (GEMM1 + LayerNorm + ReLU + GEMM2) -> thoughts (f32 + bf16)
// 128 threads = 4 waves, each wave owns a 16-row tile; block = 64 rows.
// ---------------------------------------------------------------------------
__global__ __launch_bounds__(128) void k_actor1(
    const float* __restrict__ obs, const unsigned short* __restrict__ w1b,
    const float* __restrict__ b1, const float* __restrict__ lng, const float* __restrict__ lnb,
    const unsigned short* __restrict__ w2b, const float* __restrict__ b2,
    float* __restrict__ tho, unsigned short* __restrict__ thob) {
  __shared__ float        ldsF[4][16 * 132];  // f32 tile, padded pitch
  __shared__ unsigned int ldsB[4][16 * 65];   // packed bf16 tile, padded pitch
  const int tid = threadIdx.x, w = tid >> 5, lane = tid & 31;
  const int idx = lane & 15, half = lane >> 4;
  const int rowbase = blockIdx.x * 64 + w * 16;

  // ---- GEMM1: h = obs @ w1.T  (K=256, 8 n-tiles) ----
  v8f acc[8];
#pragma unroll
  for (int i = 0; i < 8; ++i) acc[i] = vzero8();
  for (int kk = 0; kk < N_; kk += 32) {
    v16bf a = frag_f32(obs, N_, rowbase, kk, lane, false);
    if (kk + 32 < N_)
      __builtin_prefetch(obs + (size_t)(rowbase + idx) * N_ + kk + 32, 0, 1);
#pragma unroll
    for (int nt = 0; nt < 8; ++nt) {
      v16bf b = frag_bf16(w1b, N_, nt * 16, kk, lane);
      acc[nt] = wmma_bf16(a, b, acc[nt]);
    }
  }
#pragma unroll
  for (int nt = 0; nt < 8; ++nt)
#pragma unroll
    for (int r = 0; r < 8; ++r) {
      const int m = r + half * 8, n = idx + nt * 16;
      ldsF[w][m * 132 + n] = acc[nt][r] + b1[n];
    }
  __syncthreads();

  // ---- LayerNorm(E1=128) + ReLU, repack to bf16 (one lane per row) ----
  if (lane < 16) {
    float* rp = &ldsF[w][lane * 132];
    float s = 0.0f;
    for (int c = 0; c < 128; ++c) s += rp[c];
    const float mu = s * (1.0f / 128.0f);
    float v = 0.0f;
    for (int c = 0; c < 128; ++c) { float d = rp[c] - mu; v += d * d; }
    const float rs = rsqrtf(v * (1.0f / 128.0f) + 1e-5f);
    for (int c = 0; c < 128; c += 2) {
      float y0 = fmaxf((rp[c]     - mu) * rs * lng[c]     + lnb[c],     0.0f);
      float y1 = fmaxf((rp[c + 1] - mu) * rs * lng[c + 1] + lnb[c + 1], 0.0f);
      ldsB[w][lane * 65 + (c >> 1)] = bf16_pack(y0, y1);
    }
  }
  __syncthreads();

  // ---- GEMM2: thoughts = relu(ln(h)) @ w2.T  (K=128) ----
  v8f acc2[8];
#pragma unroll
  for (int i = 0; i < 8; ++i) acc2[i] = vzero8();
  for (int kk = 0; kk < M_; kk += 32) {
    v16bf a = frag_lds(ldsB[w], 65, kk, lane);
#pragma unroll
    for (int nt = 0; nt < 8; ++nt) {
      v16bf b = frag_bf16(w2b, M_, nt * 16, kk, lane);
      acc2[nt] = wmma_bf16(a, b, acc2[nt]);
    }
  }
#pragma unroll
  for (int nt = 0; nt < 8; ++nt)
#pragma unroll
    for (int r = 0; r < 8; ++r) {
      const int m = r + half * 8, n = idx + nt * 16;
      const size_t row = (size_t)rowbase + m;
      const float val = acc2[nt][r] + b2[n];
      tho[row * M_ + n]  = val;
      thob[row * M_ + n] = (unsigned short)bf16_1(val);
    }
}

// ---------------------------------------------------------------------------
// Kernel 2: attention MLP -> is_init (4 rows per 256-thread block, VALU)
// ---------------------------------------------------------------------------
__global__ __launch_bounds__(256) void k_att(
    const float* __restrict__ tho,
    const float* __restrict__ w1, const float* __restrict__ b1,
    const float* __restrict__ w2, const float* __restrict__ b2,
    const float* __restrict__ w3, const float* __restrict__ b3,
    unsigned int* __restrict__ initb) {
  __shared__ float h1[4][64];
  __shared__ float h2[4][64];
  const int tid = threadIdx.x, rl = tid >> 6, o = tid & 63;
  const size_t row = (size_t)blockIdx.x * 4 + rl;
  const float* x = tho + row * M_;
  float s = b1[o];
  const float* wr = w1 + o * M_;
#pragma unroll 4
  for (int k = 0; k < M_; ++k) s += wr[k] * x[k];
  h1[rl][o] = fmaxf(s, 0.0f);
  __syncthreads();
  float s2 = b2[o];
  const float* wr2 = w2 + o * 64;
#pragma unroll 4
  for (int k = 0; k < 64; ++k) s2 += wr2[k] * h1[rl][k];
  h2[rl][o] = fmaxf(s2, 0.0f);
  __syncthreads();
  if (o == 0) {
    float s3 = b3[0];
    for (int k = 0; k < 64; ++k) s3 += w3[k] * h2[rl][k];
    initb[row] = (s3 > 0.0f) ? 1u : 0u;   // sigmoid(s3) > 0.5  <=>  s3 > 0
  }
}

// ---------------------------------------------------------------------------
// Kernel 3: per-batch Gram matrix via WMMA + top-8 nearest (sorted indices)
// 1 block (128 threads, 4 waves) per batch; wave w = rows 16w..16w+15.
// ---------------------------------------------------------------------------
__global__ __launch_bounds__(128) void k_groups(
    const unsigned short* __restrict__ thob, int* __restrict__ idxout) {
  __shared__ float dotl[64 * 65];
  const int tid = threadIdx.x, w = tid >> 5, lane = tid & 31;
  const int idx = lane & 15, half = lane >> 4;
  const int rowbase = w * 16;
  const unsigned short* tb = thob + (size_t)blockIdx.x * A_ * M_;

  v8f acc[4];
#pragma unroll
  for (int i = 0; i < 4; ++i) acc[i] = vzero8();
  for (int kk = 0; kk < M_; kk += 32) {
    v16bf a = frag_bf16(tb, M_, rowbase, kk, lane);
#pragma unroll
    for (int nt = 0; nt < 4; ++nt) {
      v16bf b = frag_bf16(tb, M_, nt * 16, kk, lane);
      acc[nt] = wmma_bf16(a, b, acc[nt]);
    }
  }
#pragma unroll
  for (int nt = 0; nt < 4; ++nt)
#pragma unroll
    for (int r = 0; r < 8; ++r) {
      const int m = r + half * 8, n = idx + nt * 16;
      dotl[(rowbase + m) * 65 + n] = acc[nt][r];
    }
  __syncthreads();

  if (tid < 64) {
    const int i = tid;
    const float sqi = dotl[i * 65 + i];
    unsigned long long mask = 0ull;
    for (int k = 0; k < 8; ++k) {
      float best = 3.4e38f;
      int bj = 0;
      for (int j = 0; j < 64; ++j) {
        if ((mask >> j) & 1ull) continue;
        const float d = sqi - 2.0f * dotl[i * 65 + j] + dotl[j * 65 + j];
        if (d < best) { best = d; bj = j; }   // strict < : ties -> smallest index
      }
      mask |= 1ull << bj;
    }
    int* o = idxout + ((size_t)blockIdx.x * A_ + i) * 8;
    int c = 0;
    for (int j = 0; j < 64; ++j)
      if ((mask >> j) & 1ull) o[c++] = j;     // emit in ascending order (== jnp.sort)
  }
}

// ---------------------------------------------------------------------------
// Kernel 4: sequential gather -> bi-LSTM -> scatter, LDS-resident per batch.
// 1 block (256 threads) per batch. nt tile (64x128 f32) lives in LDS.
// ---------------------------------------------------------------------------
__global__ __launch_bounds__(256) void k_lstm(
    const float* __restrict__ tho, const int* __restrict__ idxin,
    const unsigned int* __restrict__ initb,
    const float* __restrict__ wihf, const float* __restrict__ whhf,
    const float* __restrict__ bihf, const float* __restrict__ bhhf,
    const float* __restrict__ wihb, const float* __restrict__ whhb,
    const float* __restrict__ bihb, const float* __restrict__ bhhb,
    float* __restrict__ newt) {
  __shared__ float nt[A_ * M_];        // 32 KB current thoughts state
  __shared__ float xz[2][8][256];      // precomputed x@wih.T + b, per dir/step
  __shared__ float zb[2][256];         // gate pre-activations per step
  __shared__ float hs[2][8][64];       // per-step hidden outputs
  __shared__ float hcur[2][64];
  __shared__ float bc[2][256];
  __shared__ int   idl[8];
  const int b = blockIdx.x, tid = threadIdx.x;

  for (int e = tid; e < A_ * M_; e += 256) nt[e] = tho[(size_t)b * A_ * M_ + e];
  bc[0][tid] = bihf[tid] + bhhf[tid];
  bc[1][tid] = bihb[tid] + bhhb[tid];
  __syncthreads();

  const float* wih[2] = { wihf, wihb };
  const float* whh[2] = { whhf, whhb };

  for (int i = 0; i < A_; ++i) {
    const unsigned int ini = initb[(size_t)b * A_ + i];   // block-uniform
    if (!ini) continue;
    if (tid < 8) idl[tid] = idxin[((size_t)b * A_ + i) * 8 + tid];
    __syncthreads();

    // precompute xz[d][t][g] = bias + dot(wih_row_g, x_t)   (backward: reversed seq)
#pragma unroll
    for (int d = 0; d < 2; ++d)
      for (int t = 0; t < 8; ++t) {
        const int src = (d == 0) ? idl[t] : idl[7 - t];
        const float* x = &nt[src * M_];
        const float* wr = wih[d] + tid * M_;
        float s = bc[d][tid];
#pragma unroll 4
        for (int k = 0; k < M_; ++k) s += wr[k] * x[k];
        xz[d][t][tid] = s;
      }
    if (tid < 64) { hcur[0][tid] = 0.0f; hcur[1][tid] = 0.0f; }
    __syncthreads();

    float creg = 0.0f;   // cell state for thread (d = tid>>6, g = tid&63) when tid<128
    for (int t = 0; t < 8; ++t) {
      float s0 = xz[0][t][tid];
      float s1 = xz[1][t][tid];
      const float* wr0 = whh[0] + tid * H_;
      const float* wr1 = whh[1] + tid * H_;
#pragma unroll 4
      for (int k = 0; k < H_; ++k) { s0 += wr0[k] * hcur[0][k]; s1 += wr1[k] * hcur[1][k]; }
      zb[0][tid] = s0;
      zb[1][tid] = s1;
      __syncthreads();
      if (tid < 128) {
        const int d = tid >> 6, g = tid & 63;
        const float zi = zb[d][g], zf = zb[d][64 + g], zg = zb[d][128 + g], zo = zb[d][192 + g];
        creg = sigm(zf) * creg + sigm(zi) * tanhf(zg);
        const float h = sigm(zo) * tanhf(creg);
        hcur[d][g] = h;
        hs[d][t][g] = h;
      }
      __syncthreads();
    }

    // scatter: row t gets [hf[t] | hb_at_original_pos_t] = [hs[0][t] | hs[1][7-t]]
    for (int e = tid; e < 8 * M_; e += 256) {
      const int t = e >> 7, c = e & 127;
      nt[idl[t] * M_ + c] = (c < 64) ? hs[0][t][c] : hs[1][7 - t][c - 64];
    }
    __syncthreads();
  }

  for (int e = tid; e < A_ * M_; e += 256) newt[(size_t)b * A_ * M_ + e] = nt[e];
}

// ---------------------------------------------------------------------------
// Kernel 5: actor_2  relu(concat) @ a2w1.T + b  ->  @ a2w2.T  -> tanh -> out
// ---------------------------------------------------------------------------
__global__ __launch_bounds__(128) void k_actor2(
    const float* __restrict__ tho, const float* __restrict__ newt,
    const unsigned short* __restrict__ w1b, const float* __restrict__ b1,
    const unsigned short* __restrict__ w2b, float* __restrict__ out) {
  __shared__ float        ldsF[4][16 * 132];
  __shared__ unsigned int ldsB[4][16 * 65];
  const int tid = threadIdx.x, w = tid >> 5, lane = tid & 31;
  const int idx = lane & 15, half = lane >> 4;
  const int rowbase = blockIdx.x * 64 + w * 16;

  // ---- GEMM: y = relu([tho | newt]) @ a2w1.T  (K=256) ----
  v8f acc[8];
#pragma unroll
  for (int i = 0; i < 8; ++i) acc[i] = vzero8();
  for (int kk = 0; kk < 2 * M_; kk += 32) {
    const float* src = (kk < M_) ? tho : newt;
    const int ko = (kk < M_) ? kk : (kk - M_);
    v16bf a = frag_f32(src, M_, rowbase, ko, lane, true);
    __builtin_prefetch(src + (size_t)(rowbase + idx) * M_ + ((ko + 32) & (M_ - 1)), 0, 1);
#pragma unroll
    for (int nt = 0; nt < 8; ++nt) {
      v16bf b = frag_bf16(w1b, 2 * M_, nt * 16, kk, lane);
      acc[nt] = wmma_bf16(a, b, acc[nt]);
    }
  }
#pragma unroll
  for (int nt = 0; nt < 8; ++nt)
#pragma unroll
    for (int r = 0; r < 8; ++r) {
      const int m = r + half * 8, n = idx + nt * 16;
      ldsF[w][m * 132 + n] = acc[nt][r] + b1[n];
    }
  __syncthreads();
  if (lane < 16) {                       // pack row to bf16 (no activation here)
    const float* rp = &ldsF[w][lane * 132];
    for (int c = 0; c < 128; c += 2)
      ldsB[w][lane * 65 + (c >> 1)] = bf16_pack(rp[c], rp[c + 1]);
  }
  __syncthreads();

  // ---- GEMM: action = tanh(y @ a2w2.T)  (K=128, N=64) ----
  v8f acc2[4];
#pragma unroll
  for (int i = 0; i < 4; ++i) acc2[i] = vzero8();
  for (int kk = 0; kk < M_; kk += 32) {
    v16bf a = frag_lds(ldsB[w], 65, kk, lane);
#pragma unroll
    for (int nt = 0; nt < 4; ++nt) {
      v16bf b = frag_bf16(w2b, M_, nt * 16, kk, lane);
      acc2[nt] = wmma_bf16(a, b, acc2[nt]);
    }
  }
#pragma unroll
  for (int nt = 0; nt < 4; ++nt)
#pragma unroll
    for (int r = 0; r < 8; ++r) {
      const int m = r + half * 8, n = idx + nt * 16;
      out[((size_t)rowbase + m) * ACT_ + n] = tanhf(acc2[nt][r]);
    }
}

// ---------------------------------------------------------------------------
// Workspace layout (bytes, all 256-aligned). Total ~86.4 MB.
// ---------------------------------------------------------------------------
static constexpr size_t OFF_THO  = 0;          // 65536*128 f32  = 33554432
static constexpr size_t OFF_NEWT = 33554432;   // 65536*128 f32  = 33554432
static constexpr size_t OFF_TBF  = 67108864;   // 65536*128 bf16 = 16777216
static constexpr size_t OFF_IDX  = 83886080;   // 65536*8 i32    =  2097152
static constexpr size_t OFF_INIT = 85983232;   // 65536 u32      =   262144
static constexpr size_t OFF_W1B  = 86245376;   // 32768 bf16
static constexpr size_t OFF_W2B  = 86310912;   // 16384 bf16
static constexpr size_t OFF_A2W1 = 86343680;   // 32768 bf16
static constexpr size_t OFF_A2W2 = 86409216;   //  8192 bf16

extern "C" void kernel_launch(void* const* d_in, const int* in_sizes, int n_in,
                              void* d_out, int out_size, void* d_ws, size_t ws_size,
                              hipStream_t stream) {
  (void)in_sizes; (void)n_in; (void)out_size; (void)ws_size;
  const float* obs    = (const float*)d_in[0];
  const float* a1_w1  = (const float*)d_in[1];
  const float* a1_b1  = (const float*)d_in[2];
  const float* ln_g   = (const float*)d_in[3];
  const float* ln_b   = (const float*)d_in[4];
  const float* a1_w2  = (const float*)d_in[5];
  const float* a1_b2  = (const float*)d_in[6];
  const float* att_w1 = (const float*)d_in[7];
  const float* att_b1 = (const float*)d_in[8];
  const float* att_w2 = (const float*)d_in[9];
  const float* att_b2 = (const float*)d_in[10];
  const float* att_w3 = (const float*)d_in[11];
  const float* att_b3 = (const float*)d_in[12];
  const float* wih_f  = (const float*)d_in[13];
  const float* whh_f  = (const float*)d_in[14];
  const float* bih_f  = (const float*)d_in[15];
  const float* bhh_f  = (const float*)d_in[16];
  const float* wih_b  = (const float*)d_in[17];
  const float* whh_b  = (const float*)d_in[18];
  const float* bih_b  = (const float*)d_in[19];
  const float* bhh_b  = (const float*)d_in[20];
  const float* a2_w1  = (const float*)d_in[21];
  const float* a2_b1  = (const float*)d_in[22];
  const float* a2_w2  = (const float*)d_in[23];

  char* ws = (char*)d_ws;
  float*          tho   = (float*)(ws + OFF_THO);
  float*          newt  = (float*)(ws + OFF_NEWT);
  unsigned short* thob  = (unsigned short*)(ws + OFF_TBF);
  int*            idxb  = (int*)(ws + OFF_IDX);
  unsigned int*   initb = (unsigned int*)(ws + OFF_INIT);
  unsigned short* w1b   = (unsigned short*)(ws + OFF_W1B);
  unsigned short* w2b   = (unsigned short*)(ws + OFF_W2B);
  unsigned short* a2w1b = (unsigned short*)(ws + OFF_A2W1);
  unsigned short* a2w2b = (unsigned short*)(ws + OFF_A2W2);

  // weight conversions (tiny, L2-resident afterwards)
  k_cvt<<<(32768 + 255) / 256, 256, 0, stream>>>(a1_w1, w1b, 32768);
  k_cvt<<<(16384 + 255) / 256, 256, 0, stream>>>(a1_w2, w2b, 16384);
  k_cvt<<<(32768 + 255) / 256, 256, 0, stream>>>(a2_w1, a2w1b, 32768);
  k_cvt<<<( 8192 + 255) / 256, 256, 0, stream>>>(a2_w2, a2w2b, 8192);

  k_actor1<<<ROWS_ / 64, 128, 0, stream>>>(obs, w1b, a1_b1, ln_g, ln_b, w2b, a1_b2, tho, thob);
  k_att<<<ROWS_ / 4, 256, 0, stream>>>(tho, att_w1, att_b1, att_w2, att_b2, att_w3, att_b3, initb);
  k_groups<<<B_, 128, 0, stream>>>(thob, idxb);
  k_lstm<<<B_, 256, 0, stream>>>(tho, idxb, initb,
                                 wih_f, whh_f, bih_f, bhh_f,
                                 wih_b, whh_b, bih_b, bhh_b, newt);
  k_actor2<<<ROWS_ / 64, 128, 0, stream>>>(tho, newt, a2w1b, a2_b1, a2w2b, (float*)d_out);
}
